// TNT_33835752358255
// MI455X (gfx1250) — matrix operand order repaired and checked by reading, not compiled, not run on hardware
//
#include <hip/hip_runtime.h>
#include <hip/hip_bf16.h>

// ---------------- model constants ----------------
#define B_    16
#define C_    256
#define H_    8
#define J_    24
#define P2_   81
#define R2_   3136
#define JP_   1944        // J*P2
#define ROWS_ 31104       // B*JP
#define HEATC_ 1968       // J + J*P2

typedef __attribute__((ext_vector_type(16))) _Float16 v16h;
typedef __attribute__((ext_vector_type(8)))  float    v8f;

#define WMMA_F16(a, b, c) \
  __builtin_amdgcn_wmma_f32_16x16x32_f16(false, (a), false, (b), (short)0, (c), false, false)

// ---------------- CDNA5 async global->LDS copy (ASYNCcnt path) ----------------
// VDST operand = LDS byte offset (flat LDS aperture address truncates to it),
// VADDR = 64-bit global address.  16B per lane.
static __device__ __forceinline__ void async_copy_b128(void* lds_dst, const void* gsrc) {
#if defined(__gfx1250__)
  unsigned int loff = (unsigned int)(unsigned long long)(uintptr_t)lds_dst;
  asm volatile("global_load_async_to_lds_b128 %0, %1, off"
               :: "v"(loff), "v"((unsigned long long)(uintptr_t)gsrc)
               : "memory");
#else
  *(float4*)lds_dst = *(const float4*)gsrc;
#endif
}
static __device__ __forceinline__ void wait_async_le1() {
#if defined(__gfx1250__)
  asm volatile("s_wait_asynccnt 0x1" ::: "memory");
#endif
}
static __device__ __forceinline__ void wait_async_le0() {
#if defined(__gfx1250__)
  asm volatile("s_wait_asynccnt 0x0" ::: "memory");
#endif
}

// ---------------- wave32 reductions ----------------
static __device__ __forceinline__ float wave_max(float v) {
#pragma unroll
  for (int o = 16; o > 0; o >>= 1) v = fmaxf(v, __shfl_xor(v, o, 32));
  return v;
}
static __device__ __forceinline__ float wave_sum(float v) {
#pragma unroll
  for (int o = 16; o > 0; o >>= 1) v += __shfl_xor(v, o, 32);
  return v;
}
static __device__ __forceinline__ float gelu_exact(float x) {
  return 0.5f * x * (1.0f + erff(x * 0.7071067811865475f));
}

// ---------------- WMMA fragment loaders ----------------
static __device__ __forceinline__ v16h frag_a_h(const _Float16* base, int stride, int lane) {
  int row = lane & 15, kb = (lane >> 4) * 16;
  const _Float16* p = base + row * stride + kb;
  v16h a;
#pragma unroll
  for (int i = 0; i < 8; i++) { a[i] = p[i]; a[8 + i] = p[8 + i]; }
  return a;
}
static __device__ __forceinline__ v16h frag_a_f(const float* base, int stride, int lane) {
  int row = lane & 15, kb = (lane >> 4) * 16;
  const float* p = base + row * stride + kb;
  v16h a;
#pragma unroll
  for (int i = 0; i < 8; i++) { a[i] = (_Float16)p[i]; a[8 + i] = (_Float16)p[8 + i]; }
  return a;
}
static __device__ __forceinline__ v16h frag_b_strided_f(const float* p, size_t stride) {
  v16h b;
#pragma unroll
  for (int i = 0; i < 16; i++) b[i] = (_Float16)p[i * stride];
  return b;
}
static __device__ __forceinline__ v16h frag_b_contig_f(const float* p) {
  v16h b;
#pragma unroll
  for (int i = 0; i < 16; i++) b[i] = (_Float16)p[i];
  return b;
}
static __device__ __forceinline__ v16h frag_b_contig_h(const _Float16* p) {
  v16h b;
#pragma unroll
  for (int i = 0; i < 16; i++) b[i] = p[i];
  return b;
}
static __device__ __forceinline__ v16h frag_b_strided_h(const _Float16* p, int stride) {
  v16h b;
#pragma unroll
  for (int i = 0; i < 16; i++) b[i] = p[i * stride];
  return b;
}

// ================= kernel 1: heat softmax row stats (max, 1/sum) =================
__global__ __launch_bounds__(256) void k_rowstats(const float* __restrict__ hm, float2* __restrict__ stats) {
  int r = blockIdx.x * 8 + (threadIdx.x >> 5);
  int lane = threadIdx.x & 31;
  int b = r / JP_, jp = r % JP_;
  const float* p = hm + ((size_t)b * HEATC_ + J_ + jp) * R2_;
  float mx = -1e30f;
  for (int i = lane; i < R2_; i += 32) mx = fmaxf(mx, p[i]);
  mx = wave_max(mx);
  float s = 0.f;
  for (int i = lane; i < R2_; i += 32) s += __expf(p[i] - mx);
  s = wave_sum(s);
  if (lane == 0) stats[r] = make_float2(mx, 1.0f / s);
}

// ================= kernel 2: pixel_tokens = softmax(heat) @ features^T =================
// per block: 16 rows (jp) x 256 cols (C), K = 3136 (bandwidth-bound; softmax fused into A-tiles)
__global__ __launch_bounds__(256) void k_pixel_gemm(const float* __restrict__ hm,
                                                    const float* __restrict__ feat,
                                                    const float2* __restrict__ stats,
                                                    float* __restrict__ X) {
  int mtile = blockIdx.x;       // 0..121
  int b = blockIdx.y;           // 0..15
  __shared__ _Float16 sA[16 * 40];
  __shared__ float sMax[16], sInv[16];
  int tid = threadIdx.x, lane = tid & 31, wave = tid >> 5;
  if (tid < 16) {
    int jp = mtile * 16 + tid;
    if (jp < JP_) { float2 st = stats[b * JP_ + jp]; sMax[tid] = st.x; sInv[tid] = st.y; }
    else          { sMax[tid] = 0.f; sInv[tid] = 0.f; }
  }
  __syncthreads();
  const float* xb = feat + (size_t)b * C_ * R2_;
  v8f acc0 = {}, acc1 = {};
  int colA = wave * 32 + (lane & 15);
  int colB = colA + 16;
  int kb2 = (lane >> 4) * 16;
  for (int k0 = 0; k0 < R2_; k0 += 32) {
    for (int e = tid; e < 512; e += 256) {
      int r = e >> 5, kk = e & 31;
      int jp = mtile * 16 + r;
      float v = 0.f;
      if (jp < JP_)
        v = __expf(hm[((size_t)b * HEATC_ + J_ + jp) * R2_ + k0 + kk] - sMax[r]);
      sA[r * 40 + kk] = (_Float16)v;
    }
    __syncthreads();
    v16h af = frag_a_h(sA, 40, lane);
    v16h b0 = frag_b_contig_f(xb + (size_t)colA * R2_ + k0 + kb2);
    v16h b1 = frag_b_contig_f(xb + (size_t)colB * R2_ + k0 + kb2);
    acc0 = WMMA_F16(af, b0, acc0);
    acc1 = WMMA_F16(af, b1, acc1);
    __syncthreads();
  }
#pragma unroll
  for (int i = 0; i < 8; i++) {
    int r = (lane >> 4) * 8 + i;
    int jp = mtile * 16 + r;
    if (jp < JP_) {
      float scale = sInv[r];
      size_t base = ((size_t)b * JP_ + jp) * C_;
      X[base + wave * 32 + (lane & 15)]      = acc0[i] * scale;
      X[base + wave * 32 + 16 + (lane & 15)] = acc1[i] * scale;
    }
  }
}

// ================= kernel 3: LayerNorm over last dim 256 =================
__global__ __launch_bounds__(256) void k_ln(const float* __restrict__ x, const float* __restrict__ g,
                                            const float* __restrict__ bta, float* __restrict__ y, int M) {
  int row = blockIdx.x * 8 + (threadIdx.x >> 5);
  int lane = threadIdx.x & 31;
  if (row >= M) return;
  const float* xr = x + (size_t)row * C_;
  float v[8], s = 0.f;
#pragma unroll
  for (int i = 0; i < 8; i++) { v[i] = xr[lane + i * 32]; s += v[i]; }
  s = wave_sum(s);
  float mean = s * (1.0f / 256.0f);
  float vs = 0.f;
#pragma unroll
  for (int i = 0; i < 8; i++) { float d = v[i] - mean; vs += d * d; }
  vs = wave_sum(vs);
  float inv = rsqrtf(vs * (1.0f / 256.0f) + 1e-6f);
  float* yr = y + (size_t)row * C_;
#pragma unroll
  for (int i = 0; i < 8; i++) {
    int c = lane + i * 32;
    yr[c] = (v[i] - mean) * inv * g[c] + bta[c];
  }
}

// ================= kernel 4: generic WMMA GEMM, Y = act(A@W + bias) [+ res] =================
// Block: 256 threads = 8 waves; tile 32(M) x 256(N); wave w -> 2x2 16x16 subtiles at cols w*32.
// A tile double-buffered via async global->LDS (f32), converted once to f16.
// Requires M % 32 == 0, K % 32 == 0.  flags bit0: exact GELU.
__global__ __launch_bounds__(256) void k_gemm(const float* __restrict__ A, const float* __restrict__ W,
                                              const float* __restrict__ bias, const float* res,
                                              float* out, int M, int N, int K, int flags) {
  __shared__ float    sAf[2][32 * 32];
  __shared__ _Float16 sAh[32 * 36];
  int tid = threadIdx.x, lane = tid & 31, wave = tid >> 5;
  int row0 = blockIdx.x * 32;
  int c0 = blockIdx.y * 256 + wave * 32;
  bool active = c0 < N;
  int colg0 = active ? (c0 + (lane & 15)) : 0;
  int colg1 = colg0 + 16;
  int kb2 = (lane >> 4) * 16;
  v8f acc00 = {}, acc01 = {}, acc10 = {}, acc11 = {};

  // each thread owns one 16B chunk of the 32x32 f32 A tile
  int sr = tid >> 3, sc = (tid & 7) * 4;
  const float* gA = A + (size_t)(row0 + sr) * K + sc;

  async_copy_b128(&sAf[0][tid * 4], gA);          // prefetch k=0
  int cur = 0;
  for (int k0 = 0; k0 < K; k0 += 32) {
    bool hasNext = (k0 + 32) < K;
    if (hasNext) async_copy_b128(&sAf[cur ^ 1][tid * 4], gA + k0 + 32);
    if (hasNext) wait_async_le1(); else wait_async_le0();
    __syncthreads();                               // sAf[cur] visible to all waves
    for (int e = tid; e < 1024; e += 256) {        // convert f32 -> f16 once
      int r = e >> 5, kk = e & 31;
      sAh[r * 36 + kk] = (_Float16)sAf[cur][e];
    }
    __syncthreads();                               // sAh ready
    if (active) {
      v16h a0 = frag_a_h(sAh, 36, lane);
      v16h a1 = frag_a_h(sAh + 16 * 36, 36, lane);
      v16h b0 = frag_b_strided_f(W + (size_t)(k0 + kb2) * N + colg0, (size_t)N);
      v16h b1 = frag_b_strided_f(W + (size_t)(k0 + kb2) * N + colg1, (size_t)N);
      acc00 = WMMA_F16(a0, b0, acc00);
      acc01 = WMMA_F16(a0, b1, acc01);
      acc10 = WMMA_F16(a1, b0, acc10);
      acc11 = WMMA_F16(a1, b1, acc11);
      if (hasNext) __builtin_prefetch(W + (size_t)(k0 + 32 + kb2) * N + colg0, 0, 3);
    }
    __syncthreads();                               // sAh/sAf[cur] reuse fence
    cur ^= 1;
  }
  if (active) {
    float bv0 = bias ? bias[colg0] : 0.f;
    float bv1 = bias ? bias[colg1] : 0.f;
#pragma unroll
    for (int i = 0; i < 8; i++) {
      int r0 = row0 + (lane >> 4) * 8 + i;
      int r1 = r0 + 16;
      float v00 = acc00[i] + bv0, v01 = acc01[i] + bv1;
      float v10 = acc10[i] + bv0, v11 = acc11[i] + bv1;
      if (flags & 1) {
        v00 = gelu_exact(v00); v01 = gelu_exact(v01);
        v10 = gelu_exact(v10); v11 = gelu_exact(v11);
      }
      if (res) {
        v00 += res[(size_t)r0 * N + colg0]; v01 += res[(size_t)r0 * N + colg1];
        v10 += res[(size_t)r1 * N + colg0]; v11 += res[(size_t)r1 * N + colg1];
      }
      out[(size_t)r0 * N + colg0] = v00;
      out[(size_t)r0 * N + colg1] = v01;
      out[(size_t)r1 * N + colg0] = v10;
      out[(size_t)r1 * N + colg1] = v11;
    }
  }
}

// ================= kernel 5: attention (one block per (batch, head)) =================
__global__ __launch_bounds__(256) void k_attn(const float* __restrict__ qkv, const float* __restrict__ bias,
                                              float* __restrict__ outp, int N, int NPAD) {
  int bh = blockIdx.x;
  int h = bh & (H_ - 1);
  int bb = bh >> 3;
  __shared__ _Float16 sQ[96 * 32], sK[96 * 32], sV[96 * 32];
  __shared__ float sS[96 * 100];
  int tid = threadIdx.x, lane = tid & 31, wave = tid >> 5;

  for (int idx = tid; idx < NPAD * 32; idx += 256) {
    int n = idx >> 5, d = idx & 31;
    float q = 0.f, k = 0.f, v = 0.f;
    if (n < N) {
      const float* row = qkv + ((size_t)(bb * N + n)) * 768 + h * 32 + d;
      q = row[0]; k = row[256]; v = row[512];
    }
    sQ[idx] = (_Float16)q; sK[idx] = (_Float16)k; sV[idx] = (_Float16)v;
  }
  __syncthreads();

  int nt16 = NPAD >> 4;
  for (int t = wave; t < nt16 * nt16; t += 8) {
    int mt = t / nt16, nt = t % nt16;
    v16h af = frag_a_h(sQ + (mt * 16) * 32, 32, lane);
    v16h bf = frag_b_contig_h(sK + (nt * 16 + (lane & 15)) * 32 + (lane >> 4) * 16);
    v8f acc = {};
    acc = WMMA_F16(af, bf, acc);
#pragma unroll
    for (int i = 0; i < 8; i++) {
      int n = mt * 16 + (lane >> 4) * 8 + i;
      int m = nt * 16 + (lane & 15);
      float sc;
      if (m < N)
        sc = acc[i] * 0.17677669529663689f + ((n < N) ? bias[(size_t)h * N * N + (size_t)n * N + m] : 0.f);
      else
        sc = -1e30f;
      sS[n * 100 + m] = sc;
    }
  }
  __syncthreads();

  for (int n = wave; n < NPAD; n += 8) {
    float mx = -1e30f;
    for (int m = lane; m < NPAD; m += 32) mx = fmaxf(mx, sS[n * 100 + m]);
    mx = wave_max(mx);
    float s = 0.f;
    for (int m = lane; m < NPAD; m += 32) { float e = __expf(sS[n * 100 + m] - mx); sS[n * 100 + m] = e; s += e; }
    s = wave_sum(s);
    float inv = 1.0f / s;
    for (int m = lane; m < NPAD; m += 32) sS[n * 100 + m] *= inv;
  }
  __syncthreads();

  int tiles = nt16 * 2;
  for (int t = wave; t < tiles; t += 8) {
    int mt = t >> 1, nt = t & 1;
    v8f acc = {};
    for (int ks = 0; ks < NPAD; ks += 32) {
      v16h af = frag_a_f(sS + (mt * 16) * 100 + ks, 100, lane);
      v16h bf = frag_b_strided_h(sV + (ks + (lane >> 4) * 16) * 32 + nt * 16 + (lane & 15), 32);
      acc = WMMA_F16(af, bf, acc);
    }
#pragma unroll
    for (int i = 0; i < 8; i++) {
      int n = mt * 16 + (lane >> 4) * 8 + i;
      if (n < N)
        outp[((size_t)(bb * N + n)) * C_ + h * 32 + nt * 16 + (lane & 15)] = acc[i];
    }
  }
}

// ================= kernel 6: final transpose (B,J,C) -> (B,C,J) =================
__global__ __launch_bounds__(256) void k_out(const float* __restrict__ patch, float* __restrict__ out) {
  int bj = blockIdx.x;
  int b = bj / J_, j = bj % J_;
  int c = threadIdx.x;
  out[((size_t)b * C_ + c) * J_ + j] = patch[(size_t)bj * C_ + c];
}

// ================= host driver =================
extern "C" void kernel_launch(void* const* d_in, const int* in_sizes, int n_in,
                              void* d_out, int out_size, void* d_ws, size_t ws_size,
                              hipStream_t stream) {
  (void)in_sizes; (void)n_in; (void)out_size; (void)ws_size;
  const float* F  = (const float*)d_in[0];
  const float* HM = (const float*)d_in[1];
  auto P = [&](int i) { return (const float*)d_in[i]; };
  // JAX sorted-dict leaf order:
  //  inner[i] @ 2+13i : bias, fc1.b, fc1.w, fc2.b, fc2.w, ln1_b, ln1_g, ln2_b, ln2_g, proj.b, proj.w, qkv.b, qkv.w
  //  outer[i] @ 54+13i (same), pd[i] @ 106+4i : l1.b,l1.w,l2.b,l2.w ; pd0 @ 122 ; pp[i] @ 126+2i : b,w ; pp0 @ 134

  float* ws = (float*)d_ws;
  size_t o = 0;
  float2* STATS = (float2*)(ws + o); o += (size_t)ROWS_ * 2;
  float* X    = ws + o; o += (size_t)ROWS_ * 256;
  float* T1   = ws + o; o += (size_t)ROWS_ * 256;
  float* QKV  = ws + o; o += (size_t)ROWS_ * 768;   // reused as MLP hidden
  float* O    = ws + o; o += (size_t)ROWS_ * 256;
  float* PD1  = ws + o; o += (size_t)ROWS_ * 64;
  float* PD2  = ws + o; o += (size_t)ROWS_ * 64;
  float* PATCH = ws + o; o += 384 * 256;
  float* PT1   = ws + o; o += 384 * 256;
  float* PQKV  = ws + o; o += 384 * 768;            // reused as patch MLP hidden
  float* PO    = ws + o; o += 384 * 256;
  float* HID  = QKV;
  float* PHID = PQKV;

  auto gemm = [&](const float* A, const float* W, const float* bias, const float* res,
                  float* out, int M, int N, int K, int flags) {
    k_gemm<<<dim3(M / 32, (N + 255) / 256), 256, 0, stream>>>(A, W, bias, res, out, M, N, K, flags);
  };

  k_rowstats<<<ROWS_ / 8, 256, 0, stream>>>(HM, STATS);
  k_pixel_gemm<<<dim3(122, B_), 256, 0, stream>>>(HM, F, STATS, X);

  gemm(X,   P(123), P(122), nullptr, PD1, ROWS_, 64, 256, 1);
  gemm(PD1, P(125), P(124), nullptr, PD2, ROWS_, 64, 64, 0);
  gemm(PD2, P(135), P(134), nullptr, PATCH, 384, 256, 5184, 0);

  for (int i = 0; i < 4; i++) {
    int bi = 2 + 13 * i, bo = 54 + 13 * i, pdi = 106 + 4 * i, ppi = 126 + 2 * i;

    // ---- inner transformer block on pixel tokens (384 x 81 x 256) ----
    k_ln<<<ROWS_ / 8, 256, 0, stream>>>(X, P(bi + 6), P(bi + 5), T1, ROWS_);
    gemm(T1, P(bi + 12), P(bi + 11), nullptr, QKV, ROWS_, 768, 256, 0);
    k_attn<<<384 * H_, 256, 0, stream>>>(QKV, P(bi + 0), O, 81, 96);
    gemm(O, P(bi + 10), P(bi + 9), X, X, ROWS_, 256, 256, 0);
    k_ln<<<ROWS_ / 8, 256, 0, stream>>>(X, P(bi + 8), P(bi + 7), T1, ROWS_);
    gemm(T1, P(bi + 2), P(bi + 1), nullptr, HID, ROWS_, 512, 256, 1);
    gemm(HID, P(bi + 4), P(bi + 3), X, X, ROWS_, 256, 512, 0);

    // ---- pixel_down + patch proj (residual into PATCH) ----
    gemm(X,   P(pdi + 1), P(pdi + 0), nullptr, PD1, ROWS_, 64, 256, 1);
    gemm(PD1, P(pdi + 3), P(pdi + 2), nullptr, PD2, ROWS_, 64, 64, 0);
    gemm(PD2, P(ppi + 1), P(ppi + 0), PATCH, PATCH, 384, 256, 5184, 0);

    // ---- outer transformer block on patch tokens (16 x 24 x 256) ----
    k_ln<<<48, 256, 0, stream>>>(PATCH, P(bo + 6), P(bo + 5), PT1, 384);
    gemm(PT1, P(bo + 12), P(bo + 11), nullptr, PQKV, 384, 768, 256, 0);
    k_attn<<<B_ * H_, 256, 0, stream>>>(PQKV, P(bo + 0), PO, 24, 32);
    gemm(PO, P(bo + 10), P(bo + 9), PATCH, PATCH, 384, 256, 256, 0);
    k_ln<<<48, 256, 0, stream>>>(PATCH, P(bo + 8), P(bo + 7), PT1, 384);
    gemm(PT1, P(bo + 2), P(bo + 1), nullptr, PHID, 384, 512, 256, 1);
    gemm(PHID, P(bo + 4), P(bo + 3), PATCH, PATCH, 384, 256, 512, 0);
  }

  k_out<<<384, 256, 0, stream>>>(PATCH, (float*)d_out);
}
